// EmbeddingBlock_43456479101352
// MI455X (gfx1250) — matrix operand order repaired and checked by reading, not compiled, not run on hardware
//
#include <hip/hip_runtime.h>

#define N_NODES 10000
#define IN_CH   128
#define EMB_D   64
#define HEADS   12
#define HDIM    (HEADS * EMB_D)   /* 768 */
#define NEG_SLOPE 0.2f

typedef __attribute__((ext_vector_type(16))) __bf16 v16bf;
typedef __attribute__((ext_vector_type(8)))  __bf16 v8bf;
typedef __attribute__((ext_vector_type(8)))  float  v8f;

// ---------------------------------------------------------------------------
// helpers
// ---------------------------------------------------------------------------
__device__ __forceinline__ v16bf merge16(v8bf lo, v8bf hi) {
  v16bf r;
#pragma unroll
  for (int i = 0; i < 8; ++i) { r[i] = lo[i]; r[i + 8] = hi[i]; }
  return r;
}

// fp32 atomic max via integer ordering trick (portable codegen on gfx1250)
__device__ __forceinline__ void atomicMaxF(float* addr, float val) {
  if (val >= 0.0f) atomicMax((int*)addr, __float_as_int(val));
  else             atomicMin((unsigned int*)addr, __float_as_uint(val));
}

// ---------------------------------------------------------------------------
// prep kernels
// ---------------------------------------------------------------------------
__global__ void cast_f32_bf16(const float* __restrict__ src, __bf16* __restrict__ dst, int n) {
  int i = blockIdx.x * blockDim.x + threadIdx.x;
  if (i < n) dst[i] = (__bf16)src[i];
}

// src is [K][Ncols] f32 row-major; dst is [Ncols][K] bf16 (K contiguous per col)
__global__ void transpose_to_bf16(const float* __restrict__ src, __bf16* __restrict__ dst,
                                  int K, int Ncols) {
  int i = blockIdx.x * blockDim.x + threadIdx.x;
  if (i >= K * Ncols) return;
  int n = i / K, k = i % K;
  dst[n * K + k] = (__bf16)src[k * Ncols + n];
}

__global__ void fill_f32(float* __restrict__ p, float v, int n) {
  int i = blockIdx.x * blockDim.x + threadIdx.x;
  if (i < n) p[i] = v;
}

// ---------------------------------------------------------------------------
// GEMM1: emb = relu(x @ We + be)   [10000,128]x[128,64]
// grid.x = 625 row tiles of 16; block = 128 threads = 4 waves (one 16-col tile each)
// ---------------------------------------------------------------------------
__global__ __launch_bounds__(128) void gemm1_emb(const __bf16* __restrict__ xb,
                                                 const __bf16* __restrict__ WeT,
                                                 const float*  __restrict__ be,
                                                 float*  __restrict__ emb,
                                                 __bf16* __restrict__ embb) {
  const int wave = threadIdx.x >> 5;
  const int lane = threadIdx.x & 31;
  const int r    = lane & 15;
  const int half = lane >> 4;
  const int row0 = blockIdx.x * 16;
  const int col0 = wave * 16;
  v8f acc = {};
#pragma unroll
  for (int kk = 0; kk < IN_CH; kk += 32) {
    // A 16x32 bf16 fragment: row = lane%16, K = kk + half*16 .. +15 (contiguous)
    const __bf16* ap = xb + (row0 + r) * IN_CH + kk + half * 16;
    v16bf A = *(const v16bf*)ap;
    // B 32x16 bf16 fragment: col = lane%16, K = kk + half*16 .. +15 (contiguous in WeT)
    v16bf B = *(const v16bf*)(WeT + (col0 + r) * IN_CH + kk + half * 16);
    acc = __builtin_amdgcn_wmma_f32_16x16x32_bf16(false, A, false, B, (short)0, acc, false, false);
  }
  const int col = col0 + r;
  const float b = be[col];
#pragma unroll
  for (int j = 0; j < 8; ++j) {      // C/D: VGPR j -> row = half*8 + j, col = lane%16
    int row = row0 + half * 8 + j;
    float v = acc[j] + b;
    v = v > 0.0f ? v : 0.0f;
    emb[row * EMB_D + col]  = v;
    embb[row * EMB_D + col] = (__bf16)v;
  }
}

// ---------------------------------------------------------------------------
// GEMM2: hh = emb @ W   [10000,64]x[64,768]
// grid = (625, 6); block = 256 = 8 waves, each owns one 16-col tile of a 128-col strip
// ---------------------------------------------------------------------------
__global__ __launch_bounds__(256) void gemm2_hh(const __bf16* __restrict__ embb,
                                                const __bf16* __restrict__ WT,
                                                float* __restrict__ hh) {
  const int wave = threadIdx.x >> 5;
  const int lane = threadIdx.x & 31;
  const int r    = lane & 15;
  const int half = lane >> 4;
  const int row0 = blockIdx.x * 16;
  const int col0 = blockIdx.y * 128 + wave * 16;
  v8f acc = {};
#pragma unroll
  for (int kk = 0; kk < EMB_D; kk += 32) {
    const __bf16* ap = embb + (row0 + r) * EMB_D + kk + half * 16;
    v16bf A = *(const v16bf*)ap;
    v16bf B = *(const v16bf*)(WT + (col0 + r) * EMB_D + kk + half * 16);
    acc = __builtin_amdgcn_wmma_f32_16x16x32_bf16(false, A, false, B, (short)0, acc, false, false);
  }
  const int col = col0 + r;
#pragma unroll
  for (int j = 0; j < 8; ++j) {
    int row = row0 + half * 8 + j;
    hh[row * HDIM + col] = acc[j];
  }
}

// ---------------------------------------------------------------------------
// per-(node,head) attention coefficients: a_src = hh . att_src, a_dst = hh . att_dst
// ---------------------------------------------------------------------------
__global__ void att_coef(const float* __restrict__ hh,
                         const float* __restrict__ att_src,
                         const float* __restrict__ att_dst,
                         float* __restrict__ a_src, float* __restrict__ a_dst) {
  int idx = blockIdx.x * blockDim.x + threadIdx.x;
  if (idx >= N_NODES * HEADS) return;
  int n = idx / HEADS, h = idx % HEADS;
  const float* hp = hh + (long long)n * HDIM + h * EMB_D;
  const float* as = att_src + h * EMB_D;
  const float* ad = att_dst + h * EMB_D;
  float s = 0.0f, t = 0.0f;
#pragma unroll 8
  for (int d = 0; d < EMB_D; ++d) { float v = hp[d]; s += v * as[d]; t += v * ad[d]; }
  a_src[idx] = s; a_dst[idx] = t;
}

// ---------------------------------------------------------------------------
// per-(edge,head) leaky-relu logits + segment max (atomic)
// ---------------------------------------------------------------------------
__global__ void edge_logits(const int* __restrict__ ei, int E,
                            const float* __restrict__ a_src, const float* __restrict__ a_dst,
                            float* __restrict__ logits, float* __restrict__ emax) {
  int idx = blockIdx.x * blockDim.x + threadIdx.x;
  int EP = E + N_NODES;
  if (idx >= EP * HEADS) return;
  int e = idx / HEADS, h = idx % HEADS;
  int s, d;
  if (e < E) { s = ei[e]; d = ei[E + e]; } else { s = d = e - E; }  // self loops
  float v = a_src[s * HEADS + h] + a_dst[d * HEADS + h];
  v = v > 0.0f ? v : NEG_SLOPE * v;
  logits[idx] = v;
  atomicMaxF(&emax[d * HEADS + h], v);
}

// ---------------------------------------------------------------------------
// per-(edge,head) exp(e - max) + segment sum (atomic); overwrites logits in place
// ---------------------------------------------------------------------------
__global__ void edge_exp(const int* __restrict__ ei, int E,
                         const float* __restrict__ emax,
                         float* __restrict__ logits, float* __restrict__ denom) {
  int idx = blockIdx.x * blockDim.x + threadIdx.x;
  int EP = E + N_NODES;
  if (idx >= EP * HEADS) return;
  int e = idx / HEADS, h = idx % HEADS;
  int d;
  if (e < E) { d = ei[E + e]; } else { d = e - E; }
  float v = __expf(logits[idx] - emax[d * HEADS + h]);
  logits[idx] = v;
  atomicAdd(&denom[d * HEADS + h], v);
}

// ---------------------------------------------------------------------------
// heavy scatter: out[dst] += alpha * hh[src]  (L2-resident: hh 30.7MB, out 30.7MB << 192MB)
// one block (192 threads) per edge; float4 gather + 4 fp32 atomics per lane
// ---------------------------------------------------------------------------
__global__ __launch_bounds__(192) void scatter_msg(const int* __restrict__ ei, int E,
                                                   const float* __restrict__ hh,
                                                   const float* __restrict__ eexp,
                                                   const float* __restrict__ denom,
                                                   float* __restrict__ out) {
  int e = blockIdx.x;
  int s, d;
  if (e < E) { s = ei[e]; d = ei[E + e]; } else { s = d = e - E; }
  int t4 = threadIdx.x * 4;                 // 192 threads * 4 = 768 = HDIM
  int h  = t4 >> 6;
  float alpha = eexp[e * HEADS + h] / (denom[d * HEADS + h] + 1e-16f);
  const float4 v = *(const float4*)(hh + (long long)s * HDIM + t4);
  float* o = out + (long long)d * HDIM + t4;
  atomicAdd(o + 0, v.x * alpha);
  atomicAdd(o + 1, v.y * alpha);
  atomicAdd(o + 2, v.z * alpha);
  atomicAdd(o + 3, v.w * alpha);
}

// ---------------------------------------------------------------------------
// finalize: y = relu(emb + mean_heads(out) + bias)
// ---------------------------------------------------------------------------
__global__ void finalize(const float* __restrict__ out, const float* __restrict__ emb,
                         const float* __restrict__ bias, float* __restrict__ y) {
  int idx = blockIdx.x * blockDim.x + threadIdx.x;
  if (idx >= N_NODES * EMB_D) return;
  int n = idx / EMB_D, d = idx % EMB_D;
  const float* op = out + (long long)n * HDIM + d;
  float s = 0.0f;
#pragma unroll
  for (int h = 0; h < HEADS; ++h) s += op[h * EMB_D];
  float v = emb[idx] + (s * (1.0f / HEADS) + bias[d]);
  y[idx] = v > 0.0f ? v : 0.0f;
}

// ---------------------------------------------------------------------------
// host launcher
// ---------------------------------------------------------------------------
extern "C" void kernel_launch(void* const* d_in, const int* in_sizes, int n_in,
                              void* d_out, int out_size, void* d_ws, size_t ws_size,
                              hipStream_t stream) {
  const float* x        = (const float*)d_in[0];
  const int*   ei       = (const int*)  d_in[1];
  const float* We       = (const float*)d_in[2];
  const float* be       = (const float*)d_in[3];
  const float* W        = (const float*)d_in[4];
  const float* att_src  = (const float*)d_in[5];
  const float* att_dst  = (const float*)d_in[6];
  const float* bias     = (const float*)d_in[7];
  float* y = (float*)d_out;

  const int E  = in_sizes[1] / 2;
  const int EP = E + N_NODES;

  // workspace layout (256B-aligned sections)
  char*  base = (char*)d_ws;
  size_t off  = 0;
  auto take = [&](size_t bytes) -> char* {
    char* q = base + off;
    off = (off + bytes + 255) & ~(size_t)255;
    return q;
  };
  __bf16* xb    = (__bf16*)take((size_t)N_NODES * IN_CH * 2);
  __bf16* WeT   = (__bf16*)take((size_t)EMB_D * IN_CH * 2);        // [64][128]
  __bf16* WT    = (__bf16*)take((size_t)HDIM * EMB_D * 2);         // [768][64]
  float*  emb   = (float*) take((size_t)N_NODES * EMB_D * 4);
  __bf16* embb  = (__bf16*)take((size_t)N_NODES * EMB_D * 2);
  float*  hh    = (float*) take((size_t)N_NODES * HDIM * 4);
  float*  a_src = (float*) take((size_t)N_NODES * HEADS * 4);
  float*  a_dst = (float*) take((size_t)N_NODES * HEADS * 4);
  float*  emax  = (float*) take((size_t)N_NODES * HEADS * 4);
  float*  denom = (float*) take((size_t)N_NODES * HEADS * 4);
  float*  eexp  = (float*) take((size_t)EP * HEADS * 4);           // logits, then exp
  float*  outb  = (float*) take((size_t)N_NODES * HDIM * 4);
  (void)ws_size; (void)n_in; (void)out_size;

  // --- prep ---
  {
    int n = N_NODES * IN_CH;
    cast_f32_bf16<<<(n + 255) / 256, 256, 0, stream>>>(x, xb, n);
  }
  transpose_to_bf16<<<(IN_CH * EMB_D + 255) / 256, 256, 0, stream>>>(We, WeT, IN_CH, EMB_D);
  transpose_to_bf16<<<(EMB_D * HDIM + 255) / 256, 256, 0, stream>>>(W, WT, EMB_D, HDIM);
  fill_f32<<<(N_NODES * HEADS + 255) / 256, 256, 0, stream>>>(emax, -3.0e38f, N_NODES * HEADS);
  fill_f32<<<(N_NODES * HEADS + 255) / 256, 256, 0, stream>>>(denom, 0.0f, N_NODES * HEADS);
  fill_f32<<<(N_NODES * HDIM + 255) / 256, 256, 0, stream>>>(outb, 0.0f, N_NODES * HDIM);

  // --- WMMA GEMMs ---
  gemm1_emb<<<N_NODES / 16, 128, 0, stream>>>(xb, WeT, be, emb, embb);
  gemm2_hh<<<dim3(N_NODES / 16, HDIM / 128), 256, 0, stream>>>(embb, WT, hh);

  // --- attention softmax over incoming edges ---
  att_coef<<<(N_NODES * HEADS + 255) / 256, 256, 0, stream>>>(hh, att_src, att_dst, a_src, a_dst);
  edge_logits<<<(EP * HEADS + 255) / 256, 256, 0, stream>>>(ei, E, a_src, a_dst, eexp, emax);
  edge_exp<<<(EP * HEADS + 255) / 256, 256, 0, stream>>>(ei, E, emax, eexp, denom);

  // --- L2-resident weighted scatter ---
  scatter_msg<<<EP, 192, 0, stream>>>(ei, E, hh, eexp, denom, outb);

  // --- head mean + bias + residual relu ---
  finalize<<<(N_NODES * EMB_D + 255) / 256, 256, 0, stream>>>(outb, emb, bias, y);
}
